// Gate_77721728189051
// MI455X (gfx1250) — matrix operand order repaired and checked by reading, not compiled, not run on hardware
//
#include <hip/hip_runtime.h>
#include <hip/hip_bf16.h>

typedef __attribute__((ext_vector_type(16))) __bf16 v16bf;
typedef __attribute__((ext_vector_type(8)))  __bf16 v8bf;
typedef __attribute__((ext_vector_type(8)))  float  v8f;

#define TOKENS 16384
#define DIM    7168
#define NEXP   256
#define TOPK   8
#define BS_STR 40      // bf16 LDS row stride: 20 words -> 16-lane b128 reads
                       // cover all 64 banks exactly once
#define WS_WB_BYTES (NEXP * DIM * 2)         // 3,670,016 (multiple of 256)

// ---------------- Phase 1: convert router weight f32 -> bf16 (once) --------
__global__ __launch_bounds__(256) void cvt_w_kernel(
    const float* __restrict__ w, __bf16* __restrict__ wb)
{
    const size_t i = ((size_t)blockIdx.x * 256 + threadIdx.x) * 8;
    const float4 a = *(const float4*)(w + i);
    const float4 b = *(const float4*)(w + i + 4);
    v8bf o;
    o[0] = (__bf16)a.x; o[1] = (__bf16)a.y; o[2] = (__bf16)a.z; o[3] = (__bf16)a.w;
    o[4] = (__bf16)b.x; o[5] = (__bf16)b.y; o[6] = (__bf16)b.z; o[7] = (__bf16)b.w;
    *(v8bf*)(wb + i) = o;
}

// ---------------- Phase 2: GEMM logits = x @ W^T (bf16 WMMA) ---------------
// WG = 8 waves x 16 tokens = 128 tokens; each wave: 1 M-tile x 16 N-tiles.
// Weight K-slice (256 experts x 32 K bf16 = 16 KB) staged in ping-pong LDS,
// shared by all waves; A fragments straight from global x (read once).
__global__ __launch_bounds__(256) void gemm_kernel(
    const float* __restrict__ x, const __bf16* __restrict__ wb,
    float* __restrict__ lgout)
{
    __shared__ __bf16 bs[2][NEXP][BS_STR];   // 2 x 20 KB ping-pong

    const int tid  = threadIdx.x;
    const int lane = tid & 31;
    const int wave = tid >> 5;
    const int hlf  = lane >> 4;
    const int l15  = lane & 15;

    // A row this lane reads (A order used here, same k order as B: lanes l and l+16 share row l15,
    // covering K {0..15} and {16..31} of each 32-wide step)
    const float* xrow = x + (size_t)(blockIdx.x * 128 + wave * 16 + l15) * DIM;
    const __bf16* wrow = wb + (size_t)tid * DIM;   // expert row staged by this thread

    v8f acc[16];
    #pragma unroll
    for (int i = 0; i < 16; ++i) acc[i] = (v8f){};

    // ---- prologue: A regs + B LDS buffer 0 for kc = 0 ----
    v8f alo = *(const v8f*)(xrow + hlf * 16);
    v8f ahi = *(const v8f*)(xrow + 8 + hlf * 16);
    {
        const v8bf c0 = *(const v8bf*)(wrow);
        const v8bf c1 = *(const v8bf*)(wrow + 8);
        const v8bf c2 = *(const v8bf*)(wrow + 16);
        const v8bf c3 = *(const v8bf*)(wrow + 24);
        *(v8bf*)&bs[0][tid][0]  = c0;  *(v8bf*)&bs[0][tid][8]  = c1;
        *(v8bf*)&bs[0][tid][16] = c2;  *(v8bf*)&bs[0][tid][24] = c3;
    }
    __syncthreads();

    for (int kc = 0; kc < DIM; kc += 32) {
        const int  buf  = (kc >> 5) & 1;
        const bool more = (kc + 32) < DIM;

        // deep x prefetch (global_prefetch): 2 KB/row ahead; one 128B line
        // covers a full 32-float K-step of this row
        if (kc + 544 <= DIM) __builtin_prefetch(xrow + kc + 512, 0, 1);

        // issue next-chunk staging loads early (latency hidden under WMMAs)
        v8f  nlo = {}, nhi = {};
        v8bf c0 = {}, c1 = {}, c2 = {}, c3 = {};
        if (more) {
            nlo = *(const v8f*)(xrow + kc + 32 + hlf * 16);
            nhi = *(const v8f*)(xrow + kc + 40 + hlf * 16);
            const __bf16* wp = wrow + kc + 32;
            c0 = *(const v8bf*)(wp);      c1 = *(const v8bf*)(wp + 8);
            c2 = *(const v8bf*)(wp + 16); c3 = *(const v8bf*)(wp + 24);
        }

        // A fragment for this K-step (v_cvt_pk_bf16_f32 x8)
        v16bf a;
        #pragma unroll
        for (int i = 0; i < 8; ++i) { a[i] = (__bf16)alo[i]; a[8 + i] = (__bf16)ahi[i]; }

        // 16 N-tiles, B fragments pipelined 2 deep so each WMMA's ds_loads
        // were issued 2 tiles earlier (no dscnt==0 stalls per tile)
        v8bf p0a = *(const v8bf*)&bs[buf][l15][hlf * 16];
        v8bf p0b = *(const v8bf*)&bs[buf][l15][hlf * 16 + 8];
        v8bf p1a = *(const v8bf*)&bs[buf][16 + l15][hlf * 16];
        v8bf p1b = *(const v8bf*)&bs[buf][16 + l15][hlf * 16 + 8];
        #pragma unroll
        for (int nt = 0; nt < 16; ++nt) {
            v8bf q0 = {}, q1 = {};
            if (nt < 14) {
                q0 = *(const v8bf*)&bs[buf][(nt + 2) * 16 + l15][hlf * 16];
                q1 = *(const v8bf*)&bs[buf][(nt + 2) * 16 + l15][hlf * 16 + 8];
            }
            v16bf b;
            #pragma unroll
            for (int i = 0; i < 8; ++i) { b[i] = p0a[i]; b[8 + i] = p0b[i]; }
            acc[nt] = __builtin_amdgcn_wmma_f32_16x16x32_bf16(
                false, a, false, b, (short)0, acc[nt], false, false);
            p0a = p1a; p0b = p1b; p1a = q0; p1b = q1;
        }

        // stage next chunk into the other buffer; single barrier per step
        if (more) {
            *(v8bf*)&bs[buf ^ 1][tid][0]  = c0;  *(v8bf*)&bs[buf ^ 1][tid][8]  = c1;
            *(v8bf*)&bs[buf ^ 1][tid][16] = c2;  *(v8bf*)&bs[buf ^ 1][tid][24] = c3;
            alo = nlo; ahi = nhi;
        }
        __syncthreads();
    }

    // ---- store logits (C layout: VGPR j -> token j / j+8 per half-wave) ----
    const int tok = blockIdx.x * 128 + wave * 16 + hlf * 8;
    #pragma unroll
    for (int nt = 0; nt < 16; ++nt) {
        #pragma unroll
        for (int j = 0; j < 8; ++j)
            lgout[(size_t)(tok + j) * NEXP + nt * 16 + l15] = acc[nt][j];
    }
}

// ---------------- Phase 3: softmax + top-8 (1 token per wave) --------------
__global__ __launch_bounds__(256) void topk_kernel(
    const float* __restrict__ lg, float* __restrict__ outW, float* __restrict__ outI)
{
    const int lane = threadIdx.x & 31;
    const int wave = threadIdx.x >> 5;
    const int t    = blockIdx.x * 8 + wave;

    float v[8];
    #pragma unroll
    for (int i = 0; i < 8; ++i) v[i] = lg[(size_t)t * NEXP + lane + 32 * i];

    float mx = v[0];
    #pragma unroll
    for (int i = 1; i < 8; ++i) mx = fmaxf(mx, v[i]);
    for (int off = 16; off > 0; off >>= 1)
        mx = fmaxf(mx, __shfl_xor(mx, off, 32));

    float sm = 0.f;
    #pragma unroll
    for (int i = 0; i < 8; ++i) sm += __expf(v[i] - mx);
    for (int off = 16; off > 0; off >>= 1)
        sm += __shfl_xor(sm, off, 32);
    const float inv = 1.0f / sm;

    unsigned used = 0;
    for (int r = 0; r < TOPK; ++r) {
        float bv = -3.402823466e+38f;
        int   bi = 0x7fffffff;
        #pragma unroll
        for (int i = 0; i < 8; ++i) {
            if (!((used >> i) & 1u)) {
                const float val = v[i];
                const int   idx = lane + 32 * i;
                if (val > bv || (val == bv && idx < bi)) { bv = val; bi = idx; }
            }
        }
        for (int off = 16; off > 0; off >>= 1) {
            const float ov = __shfl_xor(bv, off, 32);
            const int   oi = __shfl_xor(bi, off, 32);
            if (ov > bv || (ov == bv && oi < bi)) { bv = ov; bi = oi; }
        }
        if ((bi & 31) == lane) used |= 1u << (bi >> 5);
        if (lane == 0) {
            const size_t o = (size_t)t * TOPK + r;
            outW[o] = __expf(bv - mx) * inv;
            outI[o] = (float)bi;
        }
    }
}

extern "C" void kernel_launch(void* const* d_in, const int* in_sizes, int n_in,
                              void* d_out, int out_size, void* d_ws, size_t ws_size,
                              hipStream_t stream) {
    (void)in_sizes; (void)n_in; (void)out_size; (void)ws_size;
    const float* x   = (const float*)d_in[0];
    const float* wgt = (const float*)d_in[1];
    __bf16* wb   = (__bf16*)d_ws;                              // 3.67 MB bf16 W
    float*  lgS  = (float*)((char*)d_ws + WS_WB_BYTES);        // 16.7 MB logits
    float*  outW = (float*)d_out;                              // [16384, 8]
    float*  outI = outW + (size_t)TOKENS * TOPK;               // [16384, 8] (as float)

    cvt_w_kernel<<<dim3((NEXP * DIM) / (256 * 8)), dim3(256), 0, stream>>>(wgt, wb);
    gemm_kernel<<<dim3(TOKENS / 128), dim3(256), 0, stream>>>(x, wb, lgS);
    topk_kernel<<<dim3(TOKENS / 8),   dim3(256), 0, stream>>>(lgS, outW, outI);
}
